// CausalSelfAttention_1683627180595
// MI455X (gfx1250) — matrix operand order repaired and checked by reading, not compiled, not run on hardware
//
#include <hip/hip_runtime.h>

// ---------------------------------------------------------------------------
// Attention-with-copy for MI455X (gfx1250, wave32, WMMA, TDM).
//   q,k,v = E @ W{q,k,v}^T            (fp32 math, bf16 stored)
//   s     = q k^T / sqrt(H)
//   alpha = (softmax(s) + c) / (1 + S*c)      [L1 renorm closed form]
//   ctx   = ((exp(s-m)@v)/Z + c*colsum(v)) / (1 + S*c)
// Two-pass flash style; softmax stats kept in the exp2 domain.
// K^T / V tiles are fetched by the Tensor Data Mover into double-buffered
// LDS, pipelined one tile ahead of the WMMA compute (TENSORcnt-tracked).
// ---------------------------------------------------------------------------

constexpr int SEQ = 8192;
constexpr int HID = 128;

typedef unsigned short u16;
typedef __attribute__((ext_vector_type(16))) __bf16 bf16x16;
typedef __attribute__((ext_vector_type(8)))  float  f32x8;
typedef __attribute__((ext_vector_type(4)))  unsigned int u32x4v;
typedef __attribute__((ext_vector_type(4)))  int i32x4v;
typedef __attribute__((ext_vector_type(8)))  int i32x8v;

union FragB { bf16x16 v; uint4 q[2]; };

#if defined(__has_builtin)
#if __has_builtin(__builtin_amdgcn_tensor_load_to_lds) && \
    __has_builtin(__builtin_amdgcn_s_wait_tensorcnt)
#define USE_TDM 1
#endif
#endif
#ifndef USE_TDM
#define USE_TDM 0
#endif

__device__ __forceinline__ u16 f2bf(float f) {
  unsigned x = __float_as_uint(f);
  x += ((x >> 16) & 1u) + 0x7fffu;   // round-to-nearest-even
  return (u16)(x >> 16);
}

__device__ __forceinline__ float fast_exp2(float x) {
#if defined(__has_builtin) && __has_builtin(__builtin_amdgcn_exp2f)
  return __builtin_amdgcn_exp2f(x);   // v_exp_f32 (TRANS, co-executes w/ WMMA)
#else
  return exp2f(x);
#endif
}

__device__ __forceinline__ float fast_rcp(float x) {
#if defined(__has_builtin) && __has_builtin(__builtin_amdgcn_rcpf)
  return __builtin_amdgcn_rcpf(x);
#else
  return 1.0f / x;
#endif
}

#if USE_TDM
// Issue one TDM 2-D tile load: tile_w x tile_h elements of 2 bytes, row
// stride `row_stride` elements, into LDS at lds_addr (bytes, row-compact).
// D# layout per CDNA5 ISA ch.8 (group0: 4 dw, group1: 8 dw).
__device__ __forceinline__ void tdm_load_2d(unsigned lds_addr, const void* gptr,
                                            unsigned tile_w, unsigned tile_h,
                                            unsigned tensor_w, unsigned tensor_h,
                                            unsigned row_stride) {
  const unsigned long long ga = (unsigned long long)(uintptr_t)gptr;
  u32x4v g0;
  g0[0] = 1u;                                        // count=1, load, no gather
  g0[1] = lds_addr;                                  // LDS byte address
  g0[2] = (unsigned)(ga & 0xffffffffu);              // global addr [31:0]
  g0[3] = (unsigned)((ga >> 32) & 0x01ffffffu)       // global addr [56:32]
        | (2u << 30);                                // type = 2 (image)
  i32x8v g1;
  g1[0] = (int)(1u << 16);                           // data_size=1 (2B), wg_mask=0
  g1[1] = (int)((tensor_w & 0xffffu) << 16);         // tensor_dim0[15:0]
  g1[2] = (int)((tensor_w >> 16) | ((tensor_h & 0xffffu) << 16));
  g1[3] = (int)((tensor_h >> 16) | ((tile_w & 0xffffu) << 16));
  g1[4] = (int)(tile_h & 0xffffu);                   // tile_dim1 (tile_dim2=0)
  g1[5] = (int)row_stride;                           // tensor_dim0_stride[31:0]
  g1[6] = 0;                                         // stride hi, dim1_stride lo
  g1[7] = 0;
  const i32x4v z4 = {0, 0, 0, 0};
#if __clang_major__ >= 23
  const i32x8v z8 = {0, 0, 0, 0, 0, 0, 0, 0};
  __builtin_amdgcn_tensor_load_to_lds(g0, g1, z4, z4, z8, 0);
#else
  __builtin_amdgcn_tensor_load_to_lds(g0, g1, z4, z4, 0);
#endif
}

__device__ __forceinline__ unsigned lds_off(const void* p) {
  // low 32 bits of the shared-aperture flat address = LDS byte offset
  return (unsigned)(uintptr_t)p;
}
#endif  // USE_TDM

// ---------------------------------------------------------------------------
// Kernel 1: QKV projection. One block per sequence row. Writes bf16:
//   qb  [SEQ][HID] row-major, kTb [HID][SEQ] feat-major, vb [SEQ][HID].
// ---------------------------------------------------------------------------
__global__ __launch_bounds__(HID) void k_qkv(
    const float* __restrict__ E,
    const float* __restrict__ Wq, const float* __restrict__ Wk,
    const float* __restrict__ Wv,
    u16* __restrict__ qb, u16* __restrict__ kTb, u16* __restrict__ vb) {
  __shared__ float e[HID];
  const int s = blockIdx.x, t = threadIdx.x;
  e[t] = E[(size_t)s * HID + t];
  __syncthreads();
  const float* wq = Wq + t * HID;
  const float* wk = Wk + t * HID;
  const float* wv = Wv + t * HID;
  float aq = 0.f, ak = 0.f, av = 0.f;
#pragma unroll 8
  for (int i = 0; i < HID; ++i) {
    float x = e[i];
    aq = fmaf(x, wq[i], aq);
    ak = fmaf(x, wk[i], ak);
    av = fmaf(x, wv[i], av);
  }
  qb [(size_t)s * HID + t] = f2bf(aq);
  kTb[(size_t)t * SEQ + s] = f2bf(ak);
  vb [(size_t)s * HID + t] = f2bf(av);
}

// ---------------------------------------------------------------------------
// Kernel 2: vcol[h] = sum_s v[s][h] = (colsum E) @ Wv^T   (single block)
// ---------------------------------------------------------------------------
__global__ __launch_bounds__(HID) void k_vcol(
    const float* __restrict__ E, const float* __restrict__ Wv,
    float* __restrict__ vcol) {
  __shared__ float ce[HID];
  const int t = threadIdx.x;
  float s = 0.f;
  for (int r = 0; r < SEQ; ++r) s += E[(size_t)r * HID + t];
  ce[t] = s;
  __syncthreads();
  const float* w = Wv + t * HID;
  float d = 0.f;
#pragma unroll 8
  for (int i = 0; i < HID; ++i) d = fmaf(ce[i], w[i], d);
  vcol[t] = d;
}

// log2(e)/sqrt(128): scores scaled straight into the exp2 domain.
#define SC2 (0.08838834764831845f * 1.4426950408889634f)

// ---------------------------------------------------------------------------
// Kernel 3: pass 1 — per-row softmax stats (m, Z), exp2 domain.
// 256 thr = 8 waves; each wave owns 16 query rows. TDM double-buffers K^T
// tiles: DMA of tile i+1 overlaps WMMA compute on tile i; 1 barrier/iter.
// C-frag layout: VGPR g, lanes 0-15 -> row g, lanes 16-31 -> row 8+g; N=lane&15.
// ---------------------------------------------------------------------------
__global__ __launch_bounds__(256) void k_stats(
    const u16* __restrict__ qb, const u16* __restrict__ kTb,
    const int* __restrict__ mask,
    float* __restrict__ mrow, float* __restrict__ zrow) {
  __shared__ u16 sk[2][HID * 32];             // K^T tiles: [feature][32 keys] x2
  const int lane = threadIdx.x & 31;
  const int wave = threadIdx.x >> 5;
  const int half = lane >> 4;
  const int q0   = blockIdx.x * 128 + wave * 16;
  const int arow = q0 + (lane & 15);
  const int kseg = half * 8;

  FragB qa[4];                                // A frags: 16x32 bf16 q tile
#pragma unroll
  for (int j = 0; j < 4; ++j) {
    qa[j].q[0] = *reinterpret_cast<const uint4*>(qb + (size_t)arow * HID + j * 32 + kseg);
    qa[j].q[1] = *reinterpret_cast<const uint4*>(qb + (size_t)arow * HID + j * 32 + 16 + kseg);
  }

  float bias[8], M[8], Z[8];
#pragma unroll
  for (int g = 0; g < 8; ++g) {
    const int r = q0 + g + 8 * half;
    bias[g] = mask[r] ? 0.f : -1.0e30f;
    M[g] = -3.0e38f;
    Z[g] = 0.f;
  }

#if USE_TDM
  if (wave == 0)                              // prologue: DMA tile 0
    tdm_load_2d(lds_off(sk[0]), kTb, 32, HID, SEQ, HID, SEQ);
#endif

  for (int kt = 0; kt < SEQ; kt += 32) {
    const int cur = (kt >> 5) & 1;
    const u16* skc = sk[cur];
#if USE_TDM
    if (wave == 0) __builtin_amdgcn_s_wait_tensorcnt(0);   // tile kt landed
    __syncthreads();                          // visible; prev buffer consumed
    if (wave == 0 && kt + 32 < SEQ)           // DMA tile kt+32 under compute
      tdm_load_2d(lds_off(sk[cur ^ 1]), kTb + kt + 32, 32, HID, SEQ, HID, SEQ);
#else
    __syncthreads();
#pragma unroll
    for (int c = 0; c < 2; ++c) {
      const int idx = threadIdx.x + c * 256;
      const int f = idx >> 2, seg = idx & 3;
      *reinterpret_cast<uint4*>(const_cast<u16*>(skc) + f * 32 + seg * 8) =
          *reinterpret_cast<const uint4*>(kTb + (size_t)f * SEQ + kt + seg * 8);
    }
    __syncthreads();
#endif

#pragma unroll
    for (int sub = 0; sub < 2; ++sub) {
      f32x8 C = {0.f, 0.f, 0.f, 0.f, 0.f, 0.f, 0.f, 0.f};
#pragma unroll
      for (int j = 0; j < 4; ++j) {           // B frag: lane = key, elem = K feat
        FragB b;
#pragma unroll
        for (int i = 0; i < 16; ++i)
          b.v[i] = __builtin_bit_cast(__bf16, skc[(j * 32 + kseg + (i & 7) + ((i >> 3) << 4)) * 32 + sub * 16 + (lane & 15)]);
        C = __builtin_amdgcn_wmma_f32_16x16x32_bf16(false, qa[j].v, false, b.v,
                                                    (short)0, C, false, false);
      }
#pragma unroll
      for (int g = 0; g < 8; ++g) {           // branchless online (m, Z)
        const float s  = fmaf(C[g], SC2, bias[g]);
        const float nm = fmaxf(M[g], s);
        Z[g] = fmaf(Z[g], fast_exp2(M[g] - nm), fast_exp2(s - nm));
        M[g] = nm;
      }
    }
  }

  // merge the 16 lanes sharing each row (wave32 shuffles)
#pragma unroll
  for (int off = 1; off < 16; off <<= 1) {
#pragma unroll
    for (int g = 0; g < 8; ++g) {
      const float m2 = __shfl_xor(M[g], off, 32);
      const float z2 = __shfl_xor(Z[g], off, 32);
      const float nm = fmaxf(M[g], m2);
      Z[g] = fmaf(Z[g], fast_exp2(M[g] - nm), z2 * fast_exp2(m2 - nm));
      M[g] = nm;
    }
  }
  if ((lane & 15) == 0) {
#pragma unroll
    for (int g = 0; g < 8; ++g) {
      const int r = q0 + g + 8 * half;
      mrow[r] = M[g];
      zrow[r] = Z[g];
    }
  }
}

// ---------------------------------------------------------------------------
// Kernel 4: pass 2 — recompute scores, stream alphas out (non-temporal so the
// 256MB stream doesn't evict K/V from L2), accumulate ctx = exp2(P) @ V.
// K^T and V tiles double-buffered via TDM, pipelined one tile ahead.
// ---------------------------------------------------------------------------
__global__ __launch_bounds__(256) void k_ctx(
    const u16* __restrict__ qb, const u16* __restrict__ kTb,
    const u16* __restrict__ vbuf,
    const float* __restrict__ copyv, const int* __restrict__ mask,
    const float* __restrict__ mrow, const float* __restrict__ zrow,
    const float* __restrict__ vcol,
    float* __restrict__ ctx, float* __restrict__ alphas) {
  __shared__ u16 sk[2][HID * 32];            // K^T tiles [feature][32 keys] x2
  __shared__ u16 sv[2][32 * HID];            // V tiles   [32 keys][HID]     x2
  __shared__ u16 sp[8 * 16 * 32];            // per-wave exp2(P) bounce [16q][32k]
  const int lane = threadIdx.x & 31;
  const int wave = threadIdx.x >> 5;
  const int half = lane >> 4;
  const int q0   = blockIdx.x * 128 + wave * 16;
  const int arow = q0 + (lane & 15);
  const int kseg = half * 8;
  u16* myp = sp + wave * 512;

  float mg[8], a1[8], a2[8], bias[8];
#pragma unroll
  for (int g = 0; g < 8; ++g) {
    const int r = q0 + g + 8 * half;
    const float cg = copyv[r];
    const float iZ = fast_rcp(zrow[r]);
    const float iL = fast_rcp(fmaxf(1.0f + 8192.0f * cg, 1e-12f));
    mg[g]   = mrow[r];
    a1[g]   = iZ * iL;                       // alpha = p*a1 + a2
    a2[g]   = cg * iL;
    bias[g] = mask[r] ? 0.f : -1.0e30f;
  }

  FragB qa[4];
#pragma unroll
  for (int j = 0; j < 4; ++j) {
    qa[j].q[0] = *reinterpret_cast<const uint4*>(qb + (size_t)arow * HID + j * 32 + kseg);
    qa[j].q[1] = *reinterpret_cast<const uint4*>(qb + (size_t)arow * HID + j * 32 + 16 + kseg);
  }

  f32x8 acc[8];
#pragma unroll
  for (int t = 0; t < 8; ++t) acc[t] = (f32x8){0.f,0.f,0.f,0.f,0.f,0.f,0.f,0.f};

#if USE_TDM
  if (wave == 0) {                            // prologue: DMA tile 0 (K^T + V)
    tdm_load_2d(lds_off(sk[0]), kTb, 32, HID, SEQ, HID, SEQ);
    tdm_load_2d(lds_off(sv[0]), vbuf, HID, 32, HID, SEQ, HID);
  }
#endif

  for (int kt = 0; kt < SEQ; kt += 32) {
    const int cur = (kt >> 5) & 1;
    const u16* skc = sk[cur];
    const u16* svc = sv[cur];
#if USE_TDM
    if (wave == 0) __builtin_amdgcn_s_wait_tensorcnt(0);
    __syncthreads();
    if (wave == 0 && kt + 32 < SEQ) {         // DMA next tiles under compute
      tdm_load_2d(lds_off(sk[cur ^ 1]), kTb + kt + 32, 32, HID, SEQ, HID, SEQ);
      tdm_load_2d(lds_off(sv[cur ^ 1]), vbuf + (size_t)(kt + 32) * HID,
                  HID, 32, HID, SEQ, HID);
    }
#else
    __syncthreads();
#pragma unroll
    for (int c = 0; c < 2; ++c) {
      const int idx = threadIdx.x + c * 256;
      const int f = idx >> 2, seg = idx & 3;
      *reinterpret_cast<uint4*>(const_cast<u16*>(skc) + f * 32 + seg * 8) =
          *reinterpret_cast<const uint4*>(kTb + (size_t)f * SEQ + kt + seg * 8);
    }
#pragma unroll
    for (int c = 0; c < 2; ++c) {
      const int idx = threadIdx.x + c * 256;
      const int key = idx >> 4, seg = idx & 15;
      *reinterpret_cast<uint4*>(const_cast<u16*>(svc) + key * HID + seg * 8) =
          *reinterpret_cast<const uint4*>(vbuf + (size_t)(kt + key) * HID + seg * 8);
    }
    __syncthreads();
#endif

#pragma unroll
    for (int sub = 0; sub < 2; ++sub) {
      f32x8 C = {0.f, 0.f, 0.f, 0.f, 0.f, 0.f, 0.f, 0.f};
#pragma unroll
      for (int j = 0; j < 4; ++j) {
        FragB b;
#pragma unroll
        for (int i = 0; i < 16; ++i)
          b.v[i] = __builtin_bit_cast(__bf16, skc[(j * 32 + kseg + (i & 7) + ((i >> 3) << 4)) * 32 + sub * 16 + (lane & 15)]);
        C = __builtin_amdgcn_wmma_f32_16x16x32_bf16(false, qa[j].v, false, b.v,
                                                    (short)0, C, false, false);
      }
#pragma unroll
      for (int g = 0; g < 8; ++g) {
        const float p = fast_exp2(fmaf(C[g], SC2, bias[g]) - mg[g]);
        const int row = q0 + g + 8 * half;
        const int key = kt + sub * 16 + (lane & 15);
        __builtin_nontemporal_store(fmaf(p, a1[g], a2[g]),
                                    alphas + (size_t)row * SEQ + key);
        myp[(g + 8 * half) * 32 + sub * 16 + (lane & 15)] = f2bf(p);
      }
    }

    // wave-private LDS RAW: p stores above -> A-frag reads below
    asm volatile("s_wait_dscnt 0" ::: "memory");

    FragB pa;                                 // exp2(P) as 16x32 A fragment
    pa.q[0] = *reinterpret_cast<const uint4*>(myp + (lane & 15) * 32 + kseg);
    pa.q[1] = *reinterpret_cast<const uint4*>(myp + (lane & 15) * 32 + 16 + kseg);
#pragma unroll
    for (int t = 0; t < 8; ++t) {             // V B-frags: lane = h, elem = key
      FragB bv;
#pragma unroll
      for (int i = 0; i < 16; ++i)
        bv.v[i] = __builtin_bit_cast(__bf16, svc[(kseg + (i & 7) + ((i >> 3) << 4)) * HID + t * 16 + (lane & 15)]);
      acc[t] = __builtin_amdgcn_wmma_f32_16x16x32_bf16(false, pa.v, false, bv.v,
                                                       (short)0, acc[t], false, false);
    }
  }

#pragma unroll
  for (int t = 0; t < 8; ++t) {
    const float vch = vcol[t * 16 + (lane & 15)];
#pragma unroll
    for (int g = 0; g < 8; ++g) {
      const int row = q0 + g + 8 * half;
      ctx[(size_t)row * HID + t * 16 + (lane & 15)] = fmaf(acc[t][g], a1[g], a2[g] * vch);
    }
  }
}

// ---------------------------------------------------------------------------
extern "C" void kernel_launch(void* const* d_in, const int* in_sizes, int n_in,
                              void* d_out, int out_size, void* d_ws, size_t ws_size,
                              hipStream_t stream) {
  const float* E  = (const float*)d_in[0];
  const float* cp = (const float*)d_in[1];
  const float* Wq = (const float*)d_in[2];
  const float* Wk = (const float*)d_in[3];
  const float* Wv = (const float*)d_in[4];
  const int*   mk = (const int*)d_in[5];

  float* ctx    = (float*)d_out;                 // [SEQ][HID]
  float* alphas = ctx + (size_t)SEQ * HID;       // [SEQ][SEQ]

  // workspace: 3 x 2MB bf16 + stats (~6.1 MB total)
  u16* qb    = (u16*)d_ws;
  u16* kTb   = qb  + (size_t)SEQ * HID;
  u16* vb    = kTb + (size_t)SEQ * HID;
  float* mrow = (float*)(vb + (size_t)SEQ * HID);
  float* zrow = mrow + SEQ;
  float* vcol = zrow + SEQ;

  k_qkv <<<SEQ, HID, 0, stream>>>(E, Wq, Wk, Wv, qb, kTb, vb);
  k_vcol<<<1,   HID, 0, stream>>>(E, Wv, vcol);
  k_stats<<<SEQ / 128, 256, 0, stream>>>(qb, kTb, mk, mrow, zrow);
  k_ctx  <<<SEQ / 128, 256, 0, stream>>>(qb, kTb, vb, cp, mk, mrow, zrow, vcol,
                                         ctx, alphas);
}
